// MultiHeadAttention_22428319220154
// MI455X (gfx1250) — compile-verified, matrix operand order checked by reading, not run on hardware
//
#include <hip/hip_runtime.h>
#include <hip/hip_bf16.h>

typedef __attribute__((ext_vector_type(16))) _Float16 v16h;
typedef __attribute__((ext_vector_type(8)))  _Float16 v8h;
typedef __attribute__((ext_vector_type(8)))  float    v8f;

#define BB   2
#define SS   2048
#define DIMM 1024
#define HH   16
#define DD   64
#define MM   (BB * SS)   // 4096
#define KK   DIMM        // 1024

// ---------------------------------------------------------------------------
// WMMA helpers (CDNA5 16-bit fragment layout, wave32)
// A (16xK panel, row-major, leading dim ld):
//   lane<16 : row=lane,    K-halves {kb..kb+7},   {kb+16..kb+23}
//   lane>=16: row=lane-16, K-halves {kb+8..kb+15},{kb+24..kb+31}
// B fragments use the same pattern with lane = output column, reading rows of
// the K-transposed matrix (we store weights / K / V so this is contiguous).
// ---------------------------------------------------------------------------
__device__ __forceinline__ v16h load_frag16(const _Float16* base, int ld, int lane, int kb) {
  int m    = lane & 15;
  int hsel = lane >> 4;
  const _Float16* p = base + (size_t)m * ld + kb + hsel * 8;
  v8h lo = *(const v8h*)(p);
  v8h hi = *(const v8h*)(p + 16);
  union { v16h v; v8h h[2]; } u;
  u.h[0] = lo; u.h[1] = hi;
  return u.v;
}

__device__ __forceinline__ v8f wmma_f16(v16h a, v16h b, v8f c) {
  return __builtin_amdgcn_wmma_f32_16x16x32_f16(false, a, false, b, (short)0, c, false, false);
}

// ---------------------------------------------------------------------------
// Cast kernels
// ---------------------------------------------------------------------------
__global__ void cast_f32_f16(const float* __restrict__ src, _Float16* __restrict__ dst, int n) {
  int i = blockIdx.x * blockDim.x + threadIdx.x;
  if (i < n) dst[i] = (_Float16)src[i];
}

// w [K][N] f32  ->  wT [N][K] f16
__global__ void cast_weightT(const float* __restrict__ w, _Float16* __restrict__ wT) {
  int i = blockIdx.x * blockDim.x + threadIdx.x;
  if (i < KK * DIMM) {
    int k = i / DIMM, n = i % DIMM;
    wT[(size_t)n * KK + k] = (_Float16)w[i];
  }
}

// ---------------------------------------------------------------------------
// Projection GEMM: out = X[M][K] @ W[K][N] + bias  via f16 WMMA, f32 accum.
// Software-pipelined: fragments for k-step kb+32 are loaded before the WMMAs
// of step kb so load latency overlaps matrix issue.
// Block: 256 thr (8 waves). Wave -> 16x64 tile. Block -> 128x64.
// mode 0: store f16 head-split [B,H,S,D]   (Q, K)
// mode 1: store f16 transposed [B,H,D,S]   (V)
// mode 2: store f32 [M][N] to d_out        (O-projection)
// ---------------------------------------------------------------------------
__global__ __launch_bounds__(256) void proj_gemm(
    const _Float16* __restrict__ X,   // [M][K]
    const _Float16* __restrict__ WT,  // [N][K]
    const float*    __restrict__ bias,// [N]
    void* __restrict__ out, int mode)
{
  int w    = threadIdx.x >> 5;
  int lane = threadIdx.x & 31;
  int mrow = (blockIdx.y * 8 + w) * 16;
  int nb   = blockIdx.x * 64;

  const _Float16* Xrow = X + (size_t)mrow * KK;

  v8f acc[4] = {};
  // prologue: k-step 0
  v16h a0 = load_frag16(Xrow, KK, lane, 0);
  v16h b0[4];
#pragma unroll
  for (int nt = 0; nt < 4; ++nt)
    b0[nt] = load_frag16(WT + (size_t)(nb + nt * 16) * KK, KK, lane, 0);

#pragma unroll 4
  for (int kb = 0; kb < KK - 32; kb += 32) {
    // prefetch next k-step while current WMMAs issue
    v16h a1 = load_frag16(Xrow, KK, lane, kb + 32);
    v16h b1[4];
#pragma unroll
    for (int nt = 0; nt < 4; ++nt)
      b1[nt] = load_frag16(WT + (size_t)(nb + nt * 16) * KK, KK, lane, kb + 32);
#pragma unroll
    for (int nt = 0; nt < 4; ++nt)
      acc[nt] = wmma_f16(a0, b0[nt], acc[nt]);
    a0 = a1;
#pragma unroll
    for (int nt = 0; nt < 4; ++nt) b0[nt] = b1[nt];
  }
#pragma unroll
  for (int nt = 0; nt < 4; ++nt)
    acc[nt] = wmma_f16(a0, b0[nt], acc[nt]);

  int hsel = lane >> 4, col = lane & 15;
#pragma unroll
  for (int nt = 0; nt < 4; ++nt) {
    int n = nb + nt * 16 + col;
    float bv = bias[n];
#pragma unroll
    for (int r = 0; r < 8; ++r) {
      int m = mrow + r + 8 * hsel;
      float v = acc[nt][r] + bv;
      if (mode == 2) {
        ((float*)out)[(size_t)m * DIMM + n] = v;
      } else {
        int b = m >> 11, s = m & (SS - 1);
        int h = n >> 6,  d = n & (DD - 1);
        _Float16* o = (_Float16*)out;
        if (mode == 0)
          o[((size_t)(b * HH + h) * SS + s) * DD + d] = (_Float16)v;
        else
          o[((size_t)(b * HH + h) * DD + d) * SS + s] = (_Float16)v;
      }
    }
  }
}

// ---------------------------------------------------------------------------
// Attention: one block per (b, h, 16-query-row tile). 8 waves; wave w owns
// key slice [w*256, w*256+256). Full-row softmax (S=2048), weights -> d_out,
// PV via WMMA with P repacked through per-wave LDS staging.
// Bias reads and weight writes are non-temporal (pure 537 MB streams) so L2
// stays dedicated to the reused Q/K/V/weight tiles.
// ---------------------------------------------------------------------------
__global__ __launch_bounds__(256) void attn_kernel(
    const _Float16* __restrict__ Qh,   // [B,H,S,D]
    const _Float16* __restrict__ Kh,   // [B,H,S,D]
    const _Float16* __restrict__ Vt,   // [B,H,D,S]
    const float*    __restrict__ bias, // [B,H,S,S]
    const float*    __restrict__ pad,  // [B,S]
    float*          __restrict__ attw, // [B,H,S,S]  (d_out part 2)
    _Float16*       __restrict__ ctx)  // [B,S,DIM]
{
  int tq = blockIdx.x, h = blockIdx.y, b = blockIdx.z;
  int w    = threadIdx.x >> 5;
  int lane = threadIdx.x & 31;
  int hsel = lane >> 4, col = lane & 15;
  int s_base = w * 256;
  int t0 = tq * 16;

  __shared__ float wred[8 * 16];
  __shared__ float redv[16];
  __shared__ alignas(16) _Float16 pstage[8][16 * 32];
  __shared__ float opart[8][16][64];

  const size_t bh = (size_t)(b * HH + h);
  const _Float16* Qp = Qh + (bh * SS + t0) * DD;
  const _Float16* Kp = Kh + bh * SS * DD;
  const _Float16* Vp = Vt + bh * DD * SS;

  v16h aq0 = load_frag16(Qp, DD, lane, 0);
  v16h aq1 = load_frag16(Qp, DD, lane, 32);

  float mq[8];
#pragma unroll
  for (int r = 0; r < 8; ++r)
    mq[r] = 1.0f - pad[(size_t)b * SS + t0 + 8 * hsel + r];

  // ---- logits: 16 C-fragments (16 x 256 slice), K-tile double buffered ----
  v8f c[16];
  const float* bp = bias + (bh * SS + t0) * SS;
  v16h bk0 = load_frag16(Kp + (size_t)s_base * DD, DD, lane, 0);
  v16h bk1 = load_frag16(Kp + (size_t)s_base * DD, DD, lane, 32);
#pragma unroll
  for (int j = 0; j < 16; ++j) {
    int s0 = s_base + j * 16;
    v16h nk0 = bk0, nk1 = bk1;
    if (j < 15) {
      nk0 = load_frag16(Kp + (size_t)(s0 + 16) * DD, DD, lane, 0);
      nk1 = load_frag16(Kp + (size_t)(s0 + 16) * DD, DD, lane, 32);
    }
    v8f acc = {};
    acc = wmma_f16(aq0, bk0, acc);
    acc = wmma_f16(aq1, bk1, acc);
    float mk = 1.0f - pad[(size_t)b * SS + s0 + col];
#pragma unroll
    for (int r = 0; r < 8; ++r) {
      int row = 8 * hsel + r;
      float x = acc[r] * 0.125f +
                __builtin_nontemporal_load(bp + (size_t)row * SS + s0 + col);
      x += -1e9f * (1.0f - mq[r] * mk);
      acc[r] = x;
    }
    c[j] = acc;
    bk0 = nk0; bk1 = nk1;
  }

  // ---- row max: intra-wave shuffle tree + cross-wave LDS tree ----
  float lmax[8];
#pragma unroll
  for (int r = 0; r < 8; ++r) {
    float m = c[0][r];
#pragma unroll
    for (int j = 1; j < 16; ++j) m = fmaxf(m, c[j][r]);
    lmax[r] = m;
  }
#pragma unroll
  for (int mask = 1; mask <= 8; mask <<= 1)
#pragma unroll
    for (int r = 0; r < 8; ++r)
      lmax[r] = fmaxf(lmax[r], __shfl_xor(lmax[r], mask, 32));
  if (col == 0) {
#pragma unroll
    for (int r = 0; r < 8; ++r) wred[w * 16 + 8 * hsel + r] = lmax[r];
  }
  __syncthreads();
  if (threadIdx.x < 16) {
    float m = wred[threadIdx.x];
    for (int ww = 1; ww < 8; ++ww) m = fmaxf(m, wred[ww * 16 + threadIdx.x]);
    redv[threadIdx.x] = m;
  }
  __syncthreads();
  float rowmax[8];
#pragma unroll
  for (int r = 0; r < 8; ++r) rowmax[r] = redv[8 * hsel + r];
  __syncthreads();

  // ---- exp + row sum ----
  float lsum[8] = {0, 0, 0, 0, 0, 0, 0, 0};
#pragma unroll
  for (int j = 0; j < 16; ++j)
#pragma unroll
    for (int r = 0; r < 8; ++r) {
      float e = __expf(c[j][r] - rowmax[r]);
      c[j][r] = e;
      lsum[r] += e;
    }
#pragma unroll
  for (int mask = 1; mask <= 8; mask <<= 1)
#pragma unroll
    for (int r = 0; r < 8; ++r)
      lsum[r] += __shfl_xor(lsum[r], mask, 32);
  if (col == 0) {
#pragma unroll
    for (int r = 0; r < 8; ++r) wred[w * 16 + 8 * hsel + r] = lsum[r];
  }
  __syncthreads();
  if (threadIdx.x < 16) {
    float s = 0.0f;
    for (int ww = 0; ww < 8; ++ww) s += wred[ww * 16 + threadIdx.x];
    redv[threadIdx.x] = s;
  }
  __syncthreads();
  float rinv[8];
#pragma unroll
  for (int r = 0; r < 8; ++r) rinv[r] = 1.0f / redv[8 * hsel + r];

  // ---- normalize + stream attention weights to global (non-temporal) ----
  float* ap = attw + (bh * SS + t0) * SS;
#pragma unroll
  for (int j = 0; j < 16; ++j) {
    int s0 = s_base + j * 16;
#pragma unroll
    for (int r = 0; r < 8; ++r) {
      float p = c[j][r] * rinv[r];
      c[j][r] = p;
      __builtin_nontemporal_store(p, ap + (size_t)(8 * hsel + r) * SS + s0 + col);
    }
  }

  // ---- PV: repack P via LDS into A-fragments, WMMA against Vt ----
  v8f o[4] = {};
  v16h bv_cur[4];
#pragma unroll
  for (int dt = 0; dt < 4; ++dt)
    bv_cur[dt] = load_frag16(Vp + (size_t)(dt * 16) * SS + s_base, SS, lane, 0);

#pragma unroll
  for (int ks = 0; ks < 8; ++ks) {
    __syncthreads();
#pragma unroll
    for (int t = 0; t < 2; ++t) {
      int j = 2 * ks + t;
#pragma unroll
      for (int r = 0; r < 8; ++r)
        pstage[w][(8 * hsel + r) * 32 + t * 16 + col] = (_Float16)c[j][r];
    }
    __syncthreads();
    v16h a = load_frag16(&pstage[w][0], 32, lane, 0);
    v16h bv_nxt[4];
#pragma unroll
    for (int dt = 0; dt < 4; ++dt) bv_nxt[dt] = bv_cur[dt];
    if (ks < 7) {
      int sn = s_base + (ks + 1) * 32;
#pragma unroll
      for (int dt = 0; dt < 4; ++dt)
        bv_nxt[dt] = load_frag16(Vp + (size_t)(dt * 16) * SS + sn, SS, lane, 0);
    }
#pragma unroll
    for (int dt = 0; dt < 4; ++dt)
      o[dt] = wmma_f16(a, bv_cur[dt], o[dt]);
#pragma unroll
    for (int dt = 0; dt < 4; ++dt) bv_cur[dt] = bv_nxt[dt];
  }

  // ---- deterministic cross-wave context reduction ----
#pragma unroll
  for (int dt = 0; dt < 4; ++dt)
#pragma unroll
    for (int r = 0; r < 8; ++r)
      opart[w][8 * hsel + r][dt * 16 + col] = o[dt][r];
  __syncthreads();
  for (int i = threadIdx.x; i < 16 * 64; i += 256) {
    int row = i >> 6, d = i & 63;
    float s = 0.0f;
    for (int ww = 0; ww < 8; ++ww) s += opart[ww][row][d];
    ctx[((size_t)b * SS + t0 + row) * DIMM + h * DD + d] = (_Float16)s;
  }
}

// ---------------------------------------------------------------------------
extern "C" void kernel_launch(void* const* d_in, const int* in_sizes, int n_in,
                              void* d_out, int out_size, void* d_ws, size_t ws_size,
                              hipStream_t stream) {
  (void)in_sizes; (void)n_in; (void)out_size; (void)ws_size;

  const float* q    = (const float*)d_in[0];
  const float* k    = (const float*)d_in[1];
  const float* v    = (const float*)d_in[2];
  const float* bias = (const float*)d_in[3];
  const float* pad  = (const float*)d_in[4];
  const float* wq_w = (const float*)d_in[5];
  const float* wq_b = (const float*)d_in[6];
  const float* wk_w = (const float*)d_in[7];
  const float* wk_b = (const float*)d_in[8];
  const float* wv_w = (const float*)d_in[9];
  const float* wv_b = (const float*)d_in[10];
  const float* wo_w = (const float*)d_in[11];
  const float* wo_b = (const float*)d_in[12];

  char* ws = (char*)d_ws;
  const size_t MB = 1024 * 1024;
  _Float16* Xq  = (_Float16*)(ws + 0 * MB);   // 8 MB, reused as ctx later
  _Float16* Xk  = (_Float16*)(ws + 8 * MB);
  _Float16* Xv  = (_Float16*)(ws + 16 * MB);
  _Float16* wqT = (_Float16*)(ws + 24 * MB);
  _Float16* wkT = (_Float16*)(ws + 26 * MB);
  _Float16* wvT = (_Float16*)(ws + 28 * MB);
  _Float16* woT = (_Float16*)(ws + 30 * MB);
  _Float16* Qh  = (_Float16*)(ws + 32 * MB);
  _Float16* Kh  = (_Float16*)(ws + 40 * MB);
  _Float16* Vt  = (_Float16*)(ws + 48 * MB);
  _Float16* ctx = Xq;

  const int nElem = MM * KK;                  // 4,194,304
  const int cblk  = (nElem + 255) / 256;
  cast_f32_f16<<<cblk, 256, 0, stream>>>(q, Xq, nElem);
  cast_f32_f16<<<cblk, 256, 0, stream>>>(k, Xk, nElem);
  cast_f32_f16<<<cblk, 256, 0, stream>>>(v, Xv, nElem);

  const int wblk = (KK * DIMM + 255) / 256;
  cast_weightT<<<wblk, 256, 0, stream>>>(wq_w, wqT);
  cast_weightT<<<wblk, 256, 0, stream>>>(wk_w, wkT);
  cast_weightT<<<wblk, 256, 0, stream>>>(wv_w, wvT);
  cast_weightT<<<wblk, 256, 0, stream>>>(wo_w, woT);

  dim3 ggrid(DIMM / 64, MM / 128, 1);         // (16, 32)
  proj_gemm<<<ggrid, 256, 0, stream>>>(Xq, wqT, wq_b, Qh, 0);
  proj_gemm<<<ggrid, 256, 0, stream>>>(Xk, wkT, wk_b, Kh, 0);
  proj_gemm<<<ggrid, 256, 0, stream>>>(Xv, wvT, wv_b, Vt, 1);

  float* out_main = (float*)d_out;                          // [B,S,DIM]
  float* out_attw = out_main + (size_t)BB * SS * DIMM;      // [B,H,S,S]

  dim3 agrid(SS / 16, HH, BB);                // (128, 16, 2)
  attn_kernel<<<agrid, 256, 0, stream>>>(Qh, Kh, Vt, bias, pad, out_attw, ctx);

  proj_gemm<<<ggrid, 256, 0, stream>>>(ctx, woT, wo_b, out_main, 2);
}
